// BlockCirculantLinear_29549374996952
// MI455X (gfx1250) — matrix operand order repaired and checked by reading, not compiled, not run on hardware
//
#include <hip/hip_runtime.h>

typedef __attribute__((ext_vector_type(16))) _Float16 v16h;
typedef __attribute__((ext_vector_type(8)))  _Float16 v8h;
typedef __attribute__((ext_vector_type(8)))  float    v8f;
typedef __attribute__((ext_vector_type(4)))  float    v4f;

namespace {
constexpr int M_TOT = 8192;        // B*S
constexpr int N_TOT = 4096;        // OUT_F
constexpr int K_TOT = 4096;        // IN_F
constexpr int C_ROW = 4096;        // NB_IN*BLK floats per out-block row of circ_params
constexpr int BM = 128, BN = 128, BK = 32;
constexpr int LDA  = 40;           // halves per A row in LDS (32 + 8 pad) -> conflict-free
constexpr int BBLK = 264;          // halves per 8-column B block (8*32 + 8 pad)
}

__global__ __launch_bounds__(256)
void bcirc_wmma_gemm(const float* __restrict__ X,
                     const float* __restrict__ Cp,
                     const float* __restrict__ bias,
                     float* __restrict__ Out)
{
    __shared__ _Float16 As[BM * LDA];          // [m][k] row-major, f16
    __shared__ _Float16 Bs[(BN / 8) * BBLK];   // [(n>>3)][(n&7)*32 + k], f16

    const int t    = threadIdx.x;
    const int lane = t & 31;
    const int wave = t >> 5;          // 0..7
    const int wm   = wave >> 1;       // 0..3 -> 32-row strip
    const int wn   = wave & 1;        // 0..1 -> 64-col strip
    const int lm   = lane & 15;
    const int lh   = lane >> 4;

    const int m0 = blockIdx.y * BM;
    const int n0 = blockIdx.x * BN;

    // ---- A staging coords: thread t loads 16 fp32 of one row ----
    const int am = t & 127;
    const int ah = t >> 7;            // which 16-wide k half
    const float* aG = X + (size_t)(m0 + am) * K_TOT + ah * 16;
    _Float16*    aS = &As[am * LDA + ah * 16];

    // ---- B staging coords: 2 tasks, each = (one k row, one 8-col octet) ----
    const int k_l0 = t >> 4;          // 0..15
    const int k_l1 = k_l0 + 16;       // 16..31
    const int oct  = t & 15;          // which 8-column block
    const float* cG = Cp + (size_t)(n0 / 8 + oct) * C_ROW;
    _Float16*    bS = &Bs[oct * BBLK];

    v8f acc[2][4];
    #pragma unroll
    for (int r = 0; r < 2; ++r)
        #pragma unroll
        for (int c = 0; c < 4; ++c)
            #pragma unroll
            for (int v = 0; v < 8; ++v)
                acc[r][c][v] = 0.0f;

    for (int k0 = 0; k0 < K_TOT; k0 += BK) {
        // -------- global loads (b128) --------
        const float* ag = aG + k0;
        v4f a0 = *(const v4f*)(ag + 0);
        v4f a1 = *(const v4f*)(ag + 4);
        v4f a2 = *(const v4f*)(ag + 8);
        v4f a3 = *(const v4f*)(ag + 12);

        const float* cp0 = cG + k0 + (k_l0 & ~7);
        const float* cp1 = cG + k0 + (k_l1 & ~7);
        v4f c00 = *(const v4f*)(cp0);
        v4f c01 = *(const v4f*)(cp0 + 4);
        v4f c10 = *(const v4f*)(cp1);
        v4f c11 = *(const v4f*)(cp1 + 4);

        if (k0 + BK < K_TOT)
            __builtin_prefetch(aG + k0 + BK, 0, 1);   // global_prefetch_b8

        __syncthreads();   // WAR: previous iteration's fragment reads done

        // -------- A tile: fp32 -> f16, two 16B LDS stores --------
        v8h alo, ahi;
        #pragma unroll
        for (int i = 0; i < 4; ++i) {
            alo[i]     = (_Float16)a0[i];
            alo[i + 4] = (_Float16)a1[i];
            ahi[i]     = (_Float16)a2[i];
            ahi[i + 4] = (_Float16)a3[i];
        }
        *(v8h*)(aS + 0) = alo;
        *(v8h*)(aS + 8) = ahi;

        // -------- B tile: expand circulant via rotated store address --------
        // Want Bs[(n&7)*32 + k] = c[nb][kb][(n-k)&7]; for param j the dest
        // column within the octet is (j + k) & 7  (data index stays constant).
        #pragma unroll
        for (int j = 0; j < 8; ++j) {
            float v = (j < 4) ? c00[j & 3] : c01[j & 3];
            bS[(((j + k_l0) & 7) * 32) + k_l0] = (_Float16)v;
        }
        #pragma unroll
        for (int j = 0; j < 8; ++j) {
            float v = (j < 4) ? c10[j & 3] : c11[j & 3];
            bS[(((j + k_l1) & 7) * 32) + k_l1] = (_Float16)v;
        }

        __syncthreads();

        // -------- fragments (ISA 16-bit A/B lane layouts) + 8 WMMAs --------
        v16h afrag[2], bfrag[4];
        #pragma unroll
        for (int r = 0; r < 2; ++r) {
            const v8h* pa = (const v8h*)&As[(wm * 32 + r * 16 + lm) * LDA];
            v8h lo = pa[lh];        // K = lh*8 .. +7
            v8h hi = pa[2 + lh];    // K = 16 + lh*8 .. +7
            afrag[r] = __builtin_shufflevector(lo, hi,
                0,1,2,3,4,5,6,7,8,9,10,11,12,13,14,15);
        }
        #pragma unroll
        for (int c = 0; c < 4; ++c) {
            int n = wn * 64 + c * 16 + lm;      // column owned by this lane
            const v8h* pb = (const v8h*)&Bs[(n >> 3) * BBLK + (n & 7) * 32];
            v8h lo = pb[lh];                    // K = lh*8 .. +7
            v8h hi = pb[2 + lh];                // K = 16 + lh*8 .. +7
            bfrag[c] = __builtin_shufflevector(lo, hi,
                0,1,2,3,4,5,6,7,8,9,10,11,12,13,14,15);
        }
        #pragma unroll
        for (int r = 0; r < 2; ++r)
            #pragma unroll
            for (int c = 0; c < 4; ++c)
                acc[r][c] = __builtin_amdgcn_wmma_f32_16x16x32_f16(
                    false, afrag[r], false, bfrag[c],
                    (short)0, acc[r][c], false, false);
    }

    // -------- epilogue: C layout -> global, add bias --------
    #pragma unroll
    for (int r = 0; r < 2; ++r) {
        const int mb = m0 + wm * 32 + r * 16 + lh * 8;  // VGPR v holds row mb+v
        #pragma unroll
        for (int c = 0; c < 4; ++c) {
            const int n = n0 + wn * 64 + c * 16 + lm;
            const float bv = bias[n];
            #pragma unroll
            for (int v = 0; v < 8; ++v)
                Out[(size_t)(mb + v) * N_TOT + n] = acc[r][c][v] + bv;
        }
    }
}

extern "C" void kernel_launch(void* const* d_in, const int* in_sizes, int n_in,
                              void* d_out, int out_size, void* d_ws, size_t ws_size,
                              hipStream_t stream) {
    (void)in_sizes; (void)n_in; (void)d_ws; (void)ws_size; (void)out_size;
    const float* x    = (const float*)d_in[0];
    const float* cp   = (const float*)d_in[1];
    const float* bias = (const float*)d_in[2];
    float* out        = (float*)d_out;

    dim3 grid(N_TOT / BN, M_TOT / BM);   // (32, 64)
    dim3 block(256);                     // 8 wave32
    bcirc_wmma_gemm<<<grid, block, 0, stream>>>(x, cp, bias, out);
}
